// RelationalMemory_75986561401223
// MI455X (gfx1250) — matrix operand order repaired and checked by reading, not compiled, not run on hardware
//
#include <hip/hip_runtime.h>
#include <hip/hip_bf16.h>

// ---------------------------------------------------------------------------
// RelationalMemory forward step for MI455X (gfx1250, wave32, WMMA)
// Big GEMMs: v_wmma_f32_16x16x32_f16, async global->LDS fills, double buffer.
// ---------------------------------------------------------------------------

typedef __attribute__((ext_vector_type(16))) _Float16 v16h;
typedef __attribute__((ext_vector_type(8)))  _Float16 v8h;
typedef __attribute__((ext_vector_type(8)))  float    v8f;

#define BM_TILE 128
#define BN_TILE 128
#define BK_TILE 32
#define LDK     40   // padded LDS stride (halves); row pitch 80B keeps 16B alignment

#if defined(__gfx1250__) && __has_builtin(__builtin_amdgcn_global_load_async_to_lds_b128)
#define USE_ASYNC_LDS 1
#else
#define USE_ASYNC_LDS 0
#endif

#if USE_ASYNC_LDS
// builtin signature: (v4i32 AS1*, v4i32 AS3*, imm offset, imm cpol)
typedef int b128_t __attribute__((vector_size(16)));
typedef __attribute__((address_space(1))) b128_t* gp128;
typedef __attribute__((address_space(3))) b128_t* lp128;
#define AS1V(p) ((gp128)(p))
#define AS3V(p) ((lp128)(p))
__device__ __forceinline__ void wait_async0() {
#if __has_builtin(__builtin_amdgcn_s_wait_asynccnt)
    __builtin_amdgcn_s_wait_asynccnt(0);
#else
    asm volatile("s_wait_asynccnt 0x0" ::: "memory");
#endif
}
#endif

__device__ __forceinline__ v16h cat16(v8h lo, v8h hi) {
    return __builtin_shufflevector(lo, hi, 0,1,2,3,4,5,6,7,8,9,10,11,12,13,14,15);
}

// EPI: 0 = f32 out; 1 = f16 out; 2 = f16 out + xsum(relu(rl_w*x)) column epilogue;
//      3 = relu, write both f32 and f16.
template<int EPI>
__global__ __launch_bounds__(256)
void gemm_wmma_kernel(const _Float16* __restrict__ A,   // [Mtot,K] row-major f16
                      const _Float16* __restrict__ Bt,  // [N,K]    row-major f16 (pre-transposed)
                      const float*    __restrict__ bias,// [N]
                      float*          __restrict__ Cf,  // [Mtot,N] f32 (EPI 0,3)
                      _Float16*       __restrict__ Ch,  // [Mtot,N] f16 (EPI 1,2,3)
                      int Mtot, int K, int N,
                      long sA, long sB, long sBias, long sC,
                      const float* __restrict__ rlw,    // [N] (EPI 2)
                      float*       __restrict__ xsum,   // [nBatch,N] (EPI 2)
                      int rowsPerBatch)
{
    __shared__ __align__(16) _Float16 As[2][BM_TILE * LDK];
    __shared__ __align__(16) _Float16 Bs[2][BN_TILE * LDK];
    __shared__ float colsum[BN_TILE];

    const int tid  = threadIdx.x;
    const int lane = tid & 31;
    const int wave = tid >> 5;          // 0..7
    const int wm   = wave & 3;          // wave M tile (4 -> 128 rows)
    const int wn   = wave >> 2;         // wave N tile (2 -> 128 cols)
    const int l15  = lane & 15;
    const int halfSel = lane >> 4;      // 0/1
    const int half8   = halfSel * 8;

    const int tileM = blockIdx.y * BM_TILE;
    const int tileN = blockIdx.x * BN_TILE;
    const long zA = (long)blockIdx.z * sA;
    const long zB = (long)blockIdx.z * sB;
    const long zC = (long)blockIdx.z * sC;
    const long zE = (long)blockIdx.z * sBias;

    // accumulators, init with bias (per lane: fixed column, 8 rows)
    v8f acc[2][4];
    for (int j = 0; j < 4; ++j) {
        float bv = bias[zE + tileN + wn*64 + j*16 + l15];
        v8f c; for (int e = 0; e < 8; ++e) c[e] = bv;
        acc[0][j] = c; acc[1][j] = c;
    }

    // cooperative tile fill: thread -> one 32-byte row-half of A tile and B tile
    const int  ldr   = tid >> 1;            // 0..127 (tile row)
    const int  lseg  = (tid & 1) * 16;      // halves within the 32-wide K slab
    const bool aOk   = (tileM + ldr) < Mtot;
    const _Float16* Ag = A + zA + (long)(tileM + ldr) * K + lseg;
    const _Float16* Bg = Bt + zB + (long)(tileN + ldr) * K + lseg;
    const int ldsOff = ldr * LDK + lseg;

    if (EPI == 2) { if (tid < BN_TILE) colsum[tid] = 0.f; }

    int cur = 0;

#if USE_ASYNC_LDS
    // ---- async global->LDS double-buffered pipeline -----------------------
    {
        if (aOk) {
            __builtin_amdgcn_global_load_async_to_lds_b128(AS1V(Ag),     AS3V(&As[0][ldsOff]),     0, 0);
            __builtin_amdgcn_global_load_async_to_lds_b128(AS1V(Ag + 8), AS3V(&As[0][ldsOff + 8]), 0, 0);
        }
        __builtin_amdgcn_global_load_async_to_lds_b128(AS1V(Bg),     AS3V(&Bs[0][ldsOff]),     0, 0);
        __builtin_amdgcn_global_load_async_to_lds_b128(AS1V(Bg + 8), AS3V(&Bs[0][ldsOff + 8]), 0, 0);
    }
    for (int k0 = 0; k0 < K; k0 += BK_TILE) {
        wait_async0();          // this wave's fills of buf[cur] are in LDS
        __syncthreads();        // everyone's fills visible; buf[cur^1] reads done
        if (k0 + BK_TILE < K) {
            const int nxt = cur ^ 1;
            if (aOk) {
                __builtin_amdgcn_global_load_async_to_lds_b128(AS1V(Ag + k0 + BK_TILE),     AS3V(&As[nxt][ldsOff]),     0, 0);
                __builtin_amdgcn_global_load_async_to_lds_b128(AS1V(Ag + k0 + BK_TILE + 8), AS3V(&As[nxt][ldsOff + 8]), 0, 0);
            }
            __builtin_amdgcn_global_load_async_to_lds_b128(AS1V(Bg + k0 + BK_TILE),     AS3V(&Bs[nxt][ldsOff]),     0, 0);
            __builtin_amdgcn_global_load_async_to_lds_b128(AS1V(Bg + k0 + BK_TILE + 8), AS3V(&Bs[nxt][ldsOff + 8]), 0, 0);
        }
        // fragments + WMMA on buf[cur]
        const _Float16* asb = &As[cur][0];
        const _Float16* bsb = &Bs[cur][0];
        v16h afrag[2];
        for (int i = 0; i < 2; ++i) {
            int m = wm*32 + i*16 + l15;
            v8h lo = *(const v8h*)&asb[m * LDK + half8];
            v8h hi = *(const v8h*)&asb[m * LDK + half8 + 16];
            afrag[i] = cat16(lo, hi);
        }
        v16h bfrag[4];
        const int bks = halfSel * 8;
        for (int j = 0; j < 4; ++j) {
            int n = wn*64 + j*16 + l15;
            v8h lo = *(const v8h*)&bsb[n * LDK + bks];
            v8h hi = *(const v8h*)&bsb[n * LDK + bks + 16];
            bfrag[j] = cat16(lo, hi);
        }
        for (int i = 0; i < 2; ++i)
            for (int j = 0; j < 4; ++j)
                acc[i][j] = __builtin_amdgcn_wmma_f32_16x16x32_f16(
                    false, afrag[i], false, bfrag[j],
                    (short)0, acc[i][j], false, false);
        cur ^= 1;
    }
#else
    // ---- fallback: register-staged double-buffered pipeline ---------------
    v8h ra0 = {}, ra1 = {}, rb0, rb1;
    if (aOk) { ra0 = *(const v8h*)(Ag); ra1 = *(const v8h*)(Ag + 8); }
    rb0 = *(const v8h*)(Bg); rb1 = *(const v8h*)(Bg + 8);
    for (int k0 = 0; k0 < K; k0 += BK_TILE) {
        *(v8h*)&As[cur][ldsOff]     = ra0;
        *(v8h*)&As[cur][ldsOff + 8] = ra1;
        *(v8h*)&Bs[cur][ldsOff]     = rb0;
        *(v8h*)&Bs[cur][ldsOff + 8] = rb1;
        if (k0 + BK_TILE < K) {
            const int kn = k0 + BK_TILE;
            if (aOk) { ra0 = *(const v8h*)(Ag + kn); ra1 = *(const v8h*)(Ag + kn + 8); }
            rb0 = *(const v8h*)(Bg + kn); rb1 = *(const v8h*)(Bg + kn + 8);
        }
        __syncthreads();
        const _Float16* asb = &As[cur][0];
        const _Float16* bsb = &Bs[cur][0];
        v16h afrag[2];
        for (int i = 0; i < 2; ++i) {
            int m = wm*32 + i*16 + l15;
            v8h lo = *(const v8h*)&asb[m * LDK + half8];
            v8h hi = *(const v8h*)&asb[m * LDK + half8 + 16];
            afrag[i] = cat16(lo, hi);
        }
        v16h bfrag[4];
        const int bks = halfSel * 8;
        for (int j = 0; j < 4; ++j) {
            int n = wn*64 + j*16 + l15;
            v8h lo = *(const v8h*)&bsb[n * LDK + bks];
            v8h hi = *(const v8h*)&bsb[n * LDK + bks + 16];
            bfrag[j] = cat16(lo, hi);
        }
        for (int i = 0; i < 2; ++i)
            for (int j = 0; j < 4; ++j)
                acc[i][j] = __builtin_amdgcn_wmma_f32_16x16x32_f16(
                    false, afrag[i], false, bfrag[j],
                    (short)0, acc[i][j], false, false);
        cur ^= 1;
    }
#endif

    if (EPI == 2) __syncthreads();   // colsum zero-init visible

    // epilogue: lane holds column (tileN + wn*64 + j*16 + l15), rows half8+r
    for (int j = 0; j < 4; ++j) {
        const int  colL = wn*64 + j*16 + l15;
        const long col  = tileN + colL;
        float rw = 0.f;
        if (EPI == 2) rw = rlw[col];
        float ls = 0.f;
        for (int i = 0; i < 2; ++i) {
            for (int r = 0; r < 8; ++r) {
                long grow = (long)tileM + wm*32 + i*16 + half8 + r;
                if (grow < Mtot) {
                    float v = acc[i][j][r];
                    if (EPI == 3) v = fmaxf(v, 0.f);
                    long o = zC + grow * (long)N + col;
                    if (EPI == 0 || EPI == 3) Cf[o] = v;
                    if (EPI != 0)             Ch[o] = (_Float16)v;
                    if (EPI == 2)             ls += fmaxf(rw * v, 0.f);
                }
            }
        }
        if (EPI == 2) atomicAdd(&colsum[colL], ls);
    }
    if (EPI == 2) {
        __syncthreads();
        if (tid < BN_TILE) {
            int bI = tileM / rowsPerBatch;  // 128-row tile lies inside one batch (T=512)
            atomicAdd(&xsum[(long)bI * N + tileN + tid], colsum[tid]);
        }
    }
}

// ---------------------------------------------------------------------------
// fp32 -> f16 elementwise convert
__global__ __launch_bounds__(256)
void cvt_f16_kernel(const float* __restrict__ src, _Float16* __restrict__ dst, long n) {
    long i = (long)blockIdx.x * blockDim.x + threadIdx.x;
    if (i < n) dst[i] = (_Float16)src[i];
}

// fp32 [K,N] -> f16 [N,K] tiled transpose-convert (batched via z)
__global__ __launch_bounds__(256)
void transpose_cvt_kernel(const float* __restrict__ src, _Float16* __restrict__ dst,
                          int K, int N, long sS, long sD) {
    __shared__ float tile[32][33];
    const float* s = src + (long)blockIdx.z * sS;
    _Float16*    d = dst + (long)blockIdx.z * sD;
    const int n0 = blockIdx.x * 32, k0 = blockIdx.y * 32;
    const int tx = threadIdx.x, ty = threadIdx.y;       // 32 x 8
    for (int r = 0; r < 32; r += 8)
        tile[ty + r][tx] = s[(long)(k0 + ty + r) * N + n0 + tx];
    __syncthreads();
    for (int r = 0; r < 32; r += 8)
        d[(long)(n0 + ty + r) * K + k0 + tx] = (_Float16)tile[tx][ty + r];
}

// ---------------------------------------------------------------------------
// attention: per (b,h,s) softmax(q.k) @ v ;  T=512, d=128, no 1/sqrt(d) scaling
__global__ __launch_bounds__(128)
void attn_kernel(const _Float16* __restrict__ qh,  // [B*S, 1024] head-major cols
                 const _Float16* __restrict__ kh,  // [B*T, 1024]
                 const _Float16* __restrict__ vh,  // [B*T, 1024]
                 float* __restrict__ att)          // [B*S, 1024]
{
    const int idx = blockIdx.x;
    const int s = idx & 7, h = (idx >> 3) & 7, b = idx >> 6;
    __shared__ float qs[128];
    __shared__ float sc[512];
    __shared__ float red[128];
    const int tid = threadIdx.x;

    qs[tid] = (float)qh[((long)(b*8 + s)) * 1024 + h*128 + tid];
    __syncthreads();

    const _Float16* kb = kh + ((long)b * 512) * 1024 + h * 128;
    float loc[4];
    for (int u = 0; u < 4; ++u) {
        const int t = tid + u * 128;
        const v8h* kp = (const v8h*)(kb + (long)t * 1024);
        float a = 0.f;
        for (int c = 0; c < 16; ++c) {
            v8h kv = kp[c];
            for (int e = 0; e < 8; ++e) a += qs[c*8 + e] * (float)kv[e];
        }
        sc[t] = a; loc[u] = a;
    }
    float lm = fmaxf(fmaxf(loc[0], loc[1]), fmaxf(loc[2], loc[3]));
    red[tid] = lm; __syncthreads();
    for (int o = 64; o > 0; o >>= 1) { if (tid < o) red[tid] = fmaxf(red[tid], red[tid+o]); __syncthreads(); }
    const float mx = red[0]; __syncthreads();

    float es = 0.f;
    for (int u = 0; u < 4; ++u) {
        const int t = tid + u * 128;
        float e = expf(sc[t] - mx);
        sc[t] = e; es += e;
    }
    red[tid] = es; __syncthreads();
    for (int o = 64; o > 0; o >>= 1) { if (tid < o) red[tid] += red[tid+o]; __syncthreads(); }
    const float inv = 1.f / red[0];

    const _Float16* vb = vh + ((long)b * 512) * 1024 + h * 128;
    float a = 0.f;
    for (int t = 0; t < 512; ++t) a += sc[t] * (float)vb[(long)t * 1024 + tid];
    att[((long)(b*8 + s)) * 1024 + h*128 + tid] = a * inv;
}

// ---------------------------------------------------------------------------
// layernorm over last dim (D=1024): y = LN(a + r)*g + be ; optional f16 copy
__global__ __launch_bounds__(256)
void ln_kernel(const float* __restrict__ a, const float* __restrict__ r,
               const float* __restrict__ g, const float* __restrict__ be,
               float* __restrict__ yf, _Float16* __restrict__ yh, int D)
{
    const long row = blockIdx.x;
    const float* ap = a + row * D;
    const float* rp = r + row * D;
    __shared__ float red[256];
    const int tid = threadIdx.x;

    float v[4]; float sum = 0.f;
    for (int u = 0; u < 4; ++u) { v[u] = ap[tid + u*256] + rp[tid + u*256]; sum += v[u]; }
    red[tid] = sum; __syncthreads();
    for (int o = 128; o > 0; o >>= 1) { if (tid < o) red[tid] += red[tid+o]; __syncthreads(); }
    const float mean = red[0] / (float)D; __syncthreads();

    float q = 0.f;
    for (int u = 0; u < 4; ++u) { float d = v[u] - mean; q += d * d; }
    red[tid] = q; __syncthreads();
    for (int o = 128; o > 0; o >>= 1) { if (tid < o) red[tid] += red[tid+o]; __syncthreads(); }
    const float rstd = rsqrtf(red[0] / (float)D + 1e-5f);

    for (int u = 0; u < 4; ++u) {
        const int c = tid + u * 256;
        float y = (v[u] - mean) * rstd * g[c] + be[c];
        yf[row * D + c] = y;
        if (yh) yh[row * D + c] = (_Float16)y;
    }
}

// ---------------------------------------------------------------------------
// gates prep: xmeanh = f16(xsum / T);  tanhmemh[s][b][m] = f16(tanh(memory[b][s][m]))
__global__ __launch_bounds__(256)
void xmean_kernel(const float* __restrict__ xsum, _Float16* __restrict__ xmeanh,
                  float invT, long n) {
    long i = (long)blockIdx.x * blockDim.x + threadIdx.x;
    if (i < n) xmeanh[i] = (_Float16)(xsum[i] * invT);
}
__global__ __launch_bounds__(256)
void tanhmem_kernel(const float* __restrict__ memory, _Float16* __restrict__ tm, long n) {
    long i = (long)blockIdx.x * blockDim.x + threadIdx.x;
    if (i < n) {
        const int m = (int)(i & 1023);
        const int b = (int)((i >> 10) & 63);
        const int s = (int)(i >> 16);
        tm[i] = (_Float16)tanhf(memory[(long)b * 8192 + (long)s * 1024 + m]);
    }
}

// final gating: out = sigmoid(gm_i+gi_i+ib)*tanh(nm) + sigmoid(gm_f+gi_f+fb)*memory
__global__ __launch_bounds__(256)
void gates_kernel(const float* __restrict__ gm,   // [S][64][2048]
                  const float* __restrict__ gi,   // [64][2048]
                  const float* __restrict__ nm,   // [B*S][1024]
                  const float* __restrict__ memory,
                  const float* __restrict__ ib, const float* __restrict__ fb,
                  float* __restrict__ out, long n)
{
    long i = (long)blockIdx.x * blockDim.x + threadIdx.x;
    if (i >= n) return;
    const int m = (int)(i & 1023);
    const int s = (int)((i >> 10) & 7);
    const int b = (int)(i >> 13);
    const long gmo = ((long)s * 64 + b) * 2048;
    const long gio = (long)b * 2048;
    const float zi = gm[gmo + m]        + gi[gio + m]        + ib[0];
    const float zf = gm[gmo + 1024 + m] + gi[gio + 1024 + m] + fb[0];
    const float igv = 1.f / (1.f + expf(-zi));
    const float fgv = 1.f / (1.f + expf(-zf));
    const float nv = nm[((long)b * 8 + s) * 1024 + m];
    out[i] = igv * tanhf(nv) + fgv * memory[i];
}

// ---------------------------------------------------------------------------
extern "C" void kernel_launch(void* const* d_in, const int* in_sizes, int n_in,
                              void* d_out, int out_size, void* d_ws, size_t ws_size,
                              hipStream_t stream) {
    (void)in_sizes; (void)n_in; (void)out_size; (void)ws_size;
    const int B = 64, T = 512, DIN = 512, S = 8, M = 1024, NG = 2048;
    const int BT = B * T;   // 32768

    const float* inputs = (const float*)d_in[0];
    const float* memory = (const float*)d_in[1];
    const float* Wp = (const float*)d_in[2];  const float* bp = (const float*)d_in[3];
    const float* Wq = (const float*)d_in[4];  const float* bq = (const float*)d_in[5];
    const float* Wk = (const float*)d_in[6];  const float* bk = (const float*)d_in[7];
    const float* Wv = (const float*)d_in[8];  const float* bv = (const float*)d_in[9];
    const float* Wm = (const float*)d_in[10]; const float* bm = (const float*)d_in[11];
    const float* g1 = (const float*)d_in[12]; const float* be1 = (const float*)d_in[13];
    const float* g2 = (const float*)d_in[14]; const float* be2 = (const float*)d_in[15];
    const float* rl_w = (const float*)d_in[16];
    const float* rl_W = (const float*)d_in[17]; const float* rl_b = (const float*)d_in[18];
    const float* gl_W = (const float*)d_in[19]; const float* gl_b = (const float*)d_in[20];
    const float* ib = (const float*)d_in[21];  const float* fb = (const float*)d_in[22];
    float* out = (float*)d_out;

    char* ws = (char*)d_ws;
    size_t off = 0;
    auto alloc = [&](size_t bytes) -> void* {
        void* p = ws + off;
        off += (bytes + 255) & ~(size_t)255;
        return p;
    };
    _Float16* inputs_h = (_Float16*)alloc((size_t)BT * DIN * 2);
    _Float16* memh     = (_Float16*)alloc((size_t)B * S * M * 2);
    _Float16* WpT      = (_Float16*)alloc((size_t)M * DIN * 2);
    _Float16* WqT      = (_Float16*)alloc((size_t)M * M * 2);
    _Float16* WkT      = (_Float16*)alloc((size_t)M * M * 2);
    _Float16* WvT      = (_Float16*)alloc((size_t)M * M * 2);
    _Float16* WmT      = (_Float16*)alloc((size_t)M * M * 2);
    _Float16* rlWT     = (_Float16*)alloc((size_t)NG * M * 2);
    _Float16* glWT     = (_Float16*)alloc((size_t)S * NG * M * 2);
    _Float16* xh       = (_Float16*)alloc((size_t)BT * M * 2);
    _Float16* kh       = (_Float16*)alloc((size_t)BT * M * 2);
    _Float16* vh       = (_Float16*)alloc((size_t)BT * M * 2);
    _Float16* qh       = (_Float16*)alloc((size_t)B * S * M * 2);
    float*    xsum     = (float*)alloc((size_t)B * M * 4);
    float*    att      = (float*)alloc((size_t)B * S * M * 4);
    float*    mem1     = (float*)alloc((size_t)B * S * M * 4);
    _Float16* mem1h    = (_Float16*)alloc((size_t)B * S * M * 2);
    float*    mlpf     = (float*)alloc((size_t)B * S * M * 4);
    _Float16* mlph_a   = (_Float16*)alloc((size_t)B * S * M * 2);
    _Float16* mlph_b   = (_Float16*)alloc((size_t)B * S * M * 2);
    float*    nm       = (float*)alloc((size_t)B * S * M * 4);
    _Float16* xmeanh   = (_Float16*)alloc((size_t)B * M * 2);
    float*    gi       = (float*)alloc((size_t)B * NG * 4);
    _Float16* tmemh    = (_Float16*)alloc((size_t)S * B * M * 2);
    float*    gm       = (float*)alloc((size_t)S * B * NG * 4);

    (void)hipMemsetAsync(xsum, 0, (size_t)B * M * sizeof(float), stream);

    // ---- conversions -------------------------------------------------------
    {
        long n = (long)BT * DIN;
        cvt_f16_kernel<<<dim3((unsigned)((n + 255) / 256)), 256, 0, stream>>>(inputs, inputs_h, n);
        long n2 = (long)B * S * M;
        cvt_f16_kernel<<<dim3((unsigned)((n2 + 255) / 256)), 256, 0, stream>>>(memory, memh, n2);
    }
    dim3 tb(32, 8);
    transpose_cvt_kernel<<<dim3(M/32, DIN/32, 1), tb, 0, stream>>>(Wp, WpT, DIN, M, 0, 0);
    transpose_cvt_kernel<<<dim3(M/32, M/32, 1),  tb, 0, stream>>>(Wq, WqT, M, M, 0, 0);
    transpose_cvt_kernel<<<dim3(M/32, M/32, 1),  tb, 0, stream>>>(Wk, WkT, M, M, 0, 0);
    transpose_cvt_kernel<<<dim3(M/32, M/32, 1),  tb, 0, stream>>>(Wv, WvT, M, M, 0, 0);
    transpose_cvt_kernel<<<dim3(M/32, M/32, 1),  tb, 0, stream>>>(Wm, WmT, M, M, 0, 0);
    transpose_cvt_kernel<<<dim3(NG/32, M/32, 1), tb, 0, stream>>>(rl_W, rlWT, M, NG, 0, 0);
    transpose_cvt_kernel<<<dim3(NG/32, M/32, S), tb, 0, stream>>>(gl_W, glWT, M, NG,
                                                                  (long)M * NG, (long)NG * M);

    // ---- G1: x = inputs @ Wp + bp  (f16 out + relu(rl_w*x) row-sum) -------
    gemm_wmma_kernel<2><<<dim3(M/128, BT/128, 1), 256, 0, stream>>>(
        inputs_h, WpT, bp, nullptr, xh, BT, DIN, M, 0, 0, 0, 0, rl_w, xsum, T);

    // ---- G2/G3: k = x@Wk+bk, v = x@Wv+bv ----------------------------------
    gemm_wmma_kernel<1><<<dim3(M/128, BT/128, 1), 256, 0, stream>>>(
        xh, WkT, bk, nullptr, kh, BT, M, M, 0, 0, 0, 0, nullptr, nullptr, T);
    gemm_wmma_kernel<1><<<dim3(M/128, BT/128, 1), 256, 0, stream>>>(
        xh, WvT, bv, nullptr, vh, BT, M, M, 0, 0, 0, 0, nullptr, nullptr, T);

    // ---- G4: q = memory @ Wq + bq -----------------------------------------
    gemm_wmma_kernel<1><<<dim3(M/128, (B*S)/128, 1), 256, 0, stream>>>(
        memh, WqT, bq, nullptr, qh, B*S, M, M, 0, 0, 0, 0, nullptr, nullptr, T);

    // ---- attention ---------------------------------------------------------
    attn_kernel<<<dim3(B * 8 * S), 128, 0, stream>>>(qh, kh, vh, att);

    // ---- mem1 = LN(memory + att) ------------------------------------------
    ln_kernel<<<dim3(B * S), 256, 0, stream>>>(memory, att, g1, be1, mem1, mem1h, M);

    // ---- MLP: relu(relu(mem1@Wm+bm)@Wm+bm) --------------------------------
    gemm_wmma_kernel<3><<<dim3(M/128, (B*S)/128, 1), 256, 0, stream>>>(
        mem1h, WmT, bm, mlpf, mlph_a, B*S, M, M, 0, 0, 0, 0, nullptr, nullptr, T);
    gemm_wmma_kernel<3><<<dim3(M/128, (B*S)/128, 1), 256, 0, stream>>>(
        mlph_a, WmT, bm, mlpf, mlph_b, B*S, M, M, 0, 0, 0, 0, nullptr, nullptr, T);

    // ---- nm = LN(mem1 + mlp) ----------------------------------------------
    ln_kernel<<<dim3(B * S), 256, 0, stream>>>(mem1, mlpf, g2, be2, nm, nullptr, M);

    // ---- gates -------------------------------------------------------------
    xmean_kernel<<<dim3((B * M + 255) / 256), 256, 0, stream>>>(xsum, xmeanh, 1.f / (float)T,
                                                                (long)B * M);
    {
        long n = (long)S * B * M;
        tanhmem_kernel<<<dim3((unsigned)((n + 255) / 256)), 256, 0, stream>>>(memory, tmemh, n);
    }
    // gi = xmean @ rl_W + rl_b   (M=64, guarded)
    gemm_wmma_kernel<0><<<dim3(NG/128, 1, 1), 256, 0, stream>>>(
        xmeanh, rlWT, rl_b, gi, nullptr, B, M, NG, 0, 0, 0, 0, nullptr, nullptr, T);
    // gm[s] = tanh(memory[:,s,:]) @ gl_W[s] + gl_b[s]  (8 batched GEMMs, M=64)
    gemm_wmma_kernel<0><<<dim3(NG/128, 1, S), 256, 0, stream>>>(
        tmemh, glWT, gl_b, gm, nullptr, B, M, NG,
        (long)B * M, (long)NG * M, (long)NG, (long)B * NG, nullptr, nullptr, T);

    // ---- final gating ------------------------------------------------------
    {
        long n = (long)B * S * M;
        gates_kernel<<<dim3((unsigned)((n + 255) / 256)), 256, 0, stream>>>(
            gm, gi, nm, memory, ib, fb, out, n);
    }
}
